// EdgeGAT_20658792694008
// MI455X (gfx1250) — matrix operand order repaired and checked by reading, not compiled, not run on hardware
//
#include <hip/hip_runtime.h>
#include <math.h>

// ---------------------------------------------------------------------------
// EdgeGAT (2-layer GATConv w/ edge features) for MI455X (gfx1250).
//
// Roofline reasoning:
//  * Edge scatter/aggregate dominates: ~(E+N)*512B*2 layers ≈ 3.4 GB of
//    L2-resident traffic @ 23.3 TB/s HBM / 192MB L2 -> everything else must
//    be negligible.
//  * h = x@W (0.82 GFLOP/layer) goes through v_wmma_f32_16x16x32_bf16;
//    W is pre-packed in LDS in per-lane fragment order so each B fragment is
//    a single 32B contiguous LDS read (2x ds_load_b128), and each wave owns a
//    16x64 row-stripe so the A fragment is reused across 4 WMMAs.
//  * e@We then dot(a_e) is collapsed to ea . (We@a_e): removes a 6.8 GFLOP
//    edge GEMM and ~850MB of intermediate traffic; edge_attr read once for
//    both layers.
// ---------------------------------------------------------------------------

typedef __attribute__((ext_vector_type(16))) __bf16 v16bf;
typedef __attribute__((ext_vector_type(8)))  float  v8f;

#define HID 64
#define FE  16
#define NEG_SLOPE 0.2f

// float atomic max via integer atomics (valid when slot initialized to -inf)
__device__ __forceinline__ void atomicMaxFloat(float* addr, float v) {
    if (v >= 0.0f) atomicMax((int*)addr, __float_as_int(v));
    else           atomicMin((unsigned int*)addr, __float_as_uint(v));
}

// ------------------------------- utility -----------------------------------
__global__ void fill_kernel(float* __restrict__ p, long n, float v) {
    long i = (long)blockIdx.x * blockDim.x + threadIdx.x;
    if (i < n) p[i] = v;
}

// we_ae[k] = sum_j We[k, j] * a_e[j]   (both layers, 1 block of 64 threads, 32 used)
__global__ void we_ae_kernel(const float* __restrict__ We1, const float* __restrict__ a1,
                             const float* __restrict__ We2, const float* __restrict__ a2,
                             float* __restrict__ o1, float* __restrict__ o2) {
    int t = threadIdx.x;
    if (t < FE) {
        float s = 0.f;
        for (int j = 0; j < HID; ++j) s += We1[t * HID + j] * a1[j];
        o1[t] = s;
    } else if (t < 2 * FE) {
        int k = t - FE;
        float s = 0.f;
        for (int j = 0; j < HID; ++j) s += We2[k * HID + j] * a2[j];
        o2[k] = s;
    }
}

// deg[d] += 1 ; loop_sum[d][f] += edge_attr[e][f]   (thread per (edge,feat))
__global__ void deg_attr_kernel(const int* __restrict__ dst, const float* __restrict__ ea,
                                float* __restrict__ deg, float* __restrict__ lsum, int E_) {
    long t = (long)blockIdx.x * blockDim.x + threadIdx.x;
    if (t >= (long)E_ * FE) return;
    int e = (int)(t >> 4), f = (int)(t & 15);
    int d = dst[e];
    atomicAdd(lsum + (long)d * FE + f, ea[t]);
    if (f == 0) atomicAdd(deg + d, 1.0f);
}

// self-loop attention term: alpha_loop[i] = (loop_sum[i]/deg[i]) . we_ae  (both layers)
__global__ void loop_alpha_kernel(const float* __restrict__ deg, const float* __restrict__ lsum,
                                  const float* __restrict__ we1, const float* __restrict__ we2,
                                  float* __restrict__ al1, float* __restrict__ al2, int n) {
    __shared__ float w1[FE], w2[FE];
    if (threadIdx.x < FE) { w1[threadIdx.x] = we1[threadIdx.x]; w2[threadIdx.x] = we2[threadIdx.x]; }
    __syncthreads();
    int i = blockIdx.x * blockDim.x + threadIdx.x;
    if (i >= n) return;
    float dg  = deg[i];
    float inv = dg > 0.f ? 1.f / dg : 0.f;
    const float* ls = lsum + (long)i * FE;
    float s1 = 0.f, s2 = 0.f;
#pragma unroll
    for (int f = 0; f < FE; ++f) { float v = ls[f] * inv; s1 += v * w1[f]; s2 += v * w2[f]; }
    al1[i] = s1; al2[i] = s2;
}

// per-edge attention term for both layers in one edge_attr pass
__global__ void alpha_e_edges_kernel(const float* __restrict__ ea,
                                     const float* __restrict__ we1, const float* __restrict__ we2,
                                     float* __restrict__ a1, float* __restrict__ a2, int E_) {
    __shared__ float w1[FE], w2[FE];
    if (threadIdx.x < FE) { w1[threadIdx.x] = we1[threadIdx.x]; w2[threadIdx.x] = we2[threadIdx.x]; }
    __syncthreads();
    int e = blockIdx.x * blockDim.x + threadIdx.x;
    if (e >= E_) return;
    const float* row = ea + (long)e * FE;
    float s1 = 0.f, s2 = 0.f;
#pragma unroll
    for (int f = 0; f < FE; ++f) { float v = row[f]; s1 += v * w1[f]; s2 += v * w2[f]; }
    a1[e] = s1; a2[e] = s2;
}

// ---------------------- H = X @ W via WMMA (bf16 -> f32) --------------------
// Block = 256 threads = 8 waves; each wave owns a 16x64 row-stripe
// (rowTile = blockIdx.x*8 + wave). W is staged in LDS pre-packed in per-lane
// B-fragment order: fragment f = (ks*4 + colTile), lane ln, comp c at
// Wpack[(f*32 + ln)*16 + c], so a B fragment is ONE contiguous 32B LDS read.
// Per wave: 2 k-steps x 4 colTiles = 8 v_wmma_f32_16x16x32_bf16, with the
// A fragment loaded once per k-step (2x global float4 pairs) and reused 4x.
__global__ __launch_bounds__(256)
void gemm_bf16_wmma_kernel(const float* __restrict__ X, const float* __restrict__ Wg,
                           float* __restrict__ H, int n) {
    __shared__ __align__(32) __bf16 Wpack[2 * 4 * 32 * 16];   // 8 KB

    // stage + pack W (64x64 f32 row-major [k][col]) into fragment layout
    for (int p = threadIdx.x; p < 4096; p += 256) {
        int c   = p & 15;           // component within fragment
        int ln  = (p >> 4) & 31;    // lane
        int ct  = (p >> 9) & 3;     // colTile
        int ks  = p >> 11;          // k-step
        int col = ct * 16 + (ln & 15);
        int kk  = ks * 32 + (c >> 3) * 16 + (ln >> 4) * 8 + (c & 7);
        Wpack[p] = (__bf16)Wg[kk * HID + col];
    }
    __syncthreads();

    int wave = threadIdx.x >> 5;
    int lane = threadIdx.x & 31;
    int rowTile = blockIdx.x * 8 + wave;
    int row0 = rowTile * 16;
    if (row0 >= n) return;

    int l15    = lane & 15;
    int laneHi = lane >> 4;              // 0 or 1 (half-wave)
    int arow   = row0 + l15;
    if (arow >= n) arow = n - 1;         // clamp (N is a multiple of 16 here)

    const v16bf* bfrag = (const v16bf*)Wpack;   // 32B per fragment slot

    v8f acc[4] = {};
#pragma unroll
    for (int ks = 0; ks < 2; ++ks) {
        // A fragment (16x32 bf16): lane half selects K 0-7/16-23 vs 8-15/24-31
        const float* xr = X + (long)arow * HID + ks * 32 + laneHi * 8;
        float4 p0 = *(const float4*)(xr);
        float4 p1 = *(const float4*)(xr + 4);
        float4 p2 = *(const float4*)(xr + 16);
        float4 p3 = *(const float4*)(xr + 20);
        v16bf a;
        a[0]  = (__bf16)p0.x; a[1]  = (__bf16)p0.y; a[2]  = (__bf16)p0.z; a[3]  = (__bf16)p0.w;
        a[4]  = (__bf16)p1.x; a[5]  = (__bf16)p1.y; a[6]  = (__bf16)p1.z; a[7]  = (__bf16)p1.w;
        a[8]  = (__bf16)p2.x; a[9]  = (__bf16)p2.y; a[10] = (__bf16)p2.z; a[11] = (__bf16)p2.w;
        a[12] = (__bf16)p3.x; a[13] = (__bf16)p3.y; a[14] = (__bf16)p3.z; a[15] = (__bf16)p3.w;
#pragma unroll
        for (int ct = 0; ct < 4; ++ct) {
            v16bf b = bfrag[(ks * 4 + ct) * 32 + lane];   // 2x ds_load_b128
            acc[ct] = __builtin_amdgcn_wmma_f32_16x16x32_bf16(
                /*neg_a=*/false, a, /*neg_b=*/false, b,
                /*c_mod=*/(short)0, acc[ct], /*reuse_a=*/false, /*reuse_b=*/false);
        }
    }

    // D: VGPR r holds row M = laneHi*8 + r, column N = l15 (per colTile)
    if (row0 + 16 <= n) {                 // full tile: straight-line stores
#pragma unroll
        for (int ct = 0; ct < 4; ++ct) {
            float* hb = H + (long)(row0 + laneHi * 8) * HID + ct * 16 + l15;
#pragma unroll
            for (int r = 0; r < 8; ++r) hb[(long)r * HID] = acc[ct][r];
        }
    } else {                              // ragged tail tile
#pragma unroll
        for (int ct = 0; ct < 4; ++ct) {
            int col = ct * 16 + l15;
#pragma unroll
            for (int r = 0; r < 8; ++r) {
                int rr = row0 + laneHi * 8 + r;
                if (rr < n) H[(long)rr * HID + col] = acc[ct][r];
            }
        }
    }
}

// alpha_src[i] = h[i].a_src ; alpha_dst[i] = h[i].a_dst
__global__ void alpha_nodes_kernel(const float* __restrict__ h,
                                   const float* __restrict__ avs, const float* __restrict__ avd,
                                   float* __restrict__ oa, float* __restrict__ ob, int n) {
    __shared__ float sa[HID], sb[HID];
    if (threadIdx.x < HID) { sa[threadIdx.x] = avs[threadIdx.x]; sb[threadIdx.x] = avd[threadIdx.x]; }
    __syncthreads();
    int i = blockIdx.x * blockDim.x + threadIdx.x;
    if (i >= n) return;
    const float* hr = h + (long)i * HID;
    float da = 0.f, db = 0.f;
#pragma unroll 8
    for (int f = 0; f < HID; ++f) { float v = hr[f]; da += v * sa[f]; db += v * sb[f]; }
    oa[i] = da; ob[i] = db;
}

// leaky-relu logits for E edges + N self loops, running segment max over dst
__global__ void logits_kernel(const int* __restrict__ src, const int* __restrict__ dst,
                              const float* __restrict__ aS, const float* __restrict__ aD,
                              const float* __restrict__ aeE, const float* __restrict__ aeL,
                              float* __restrict__ lg, float* __restrict__ smax, int E_, int n) {
    long i = (long)blockIdx.x * blockDim.x + threadIdx.x;
    long M = (long)E_ + n;
    if (i >= M) return;
    int s, d; float ae;
    if (i < E_) { s = src[i]; d = dst[i]; ae = aeE[i]; }
    else        { s = d = (int)(i - E_); ae = aeL[s]; }
    float x = aS[s] + aD[d] + ae;
    float v = x > 0.f ? x : NEG_SLOPE * x;
    lg[i] = v;
    atomicMaxFloat(smax + d, v);
}

// exp(logit - seg_max) in place + segment denominator
__global__ void exp_kernel(const int* __restrict__ dst, float* __restrict__ lg,
                           const float* __restrict__ smax, float* __restrict__ den,
                           int E_, int n) {
    long i = (long)blockIdx.x * blockDim.x + threadIdx.x;
    long M = (long)E_ + n;
    if (i >= M) return;
    int d = (i < E_) ? dst[i] : (int)(i - E_);
    float e = expf(lg[i] - smax[d]);
    lg[i] = e;
    atomicAdd(den + d, e);
}

// out[dst] += h[src] * (ex/denom[dst]) — one wave32 per (edge|selfloop),
// lane covers features {lane, lane+32}: coalesced 128B gathers, f32 atomics.
__global__ __launch_bounds__(256)
void aggregate_kernel(const int* __restrict__ src, const int* __restrict__ dst,
                      const float* __restrict__ h, const float* __restrict__ ex,
                      const float* __restrict__ den, float* __restrict__ out,
                      int E_, int n) {
    long tid = (long)blockIdx.x * blockDim.x + threadIdx.x;
    long gw  = tid >> 5;                 // global wave id == edge id
    int lane = threadIdx.x & 31;
    long M = (long)E_ + n;
    if (gw >= M) return;
    int s, d;
    if (gw < E_) {
        s = src[gw]; d = dst[gw];
        if (gw + 2048 < E_) __builtin_prefetch(src + gw + 2048, 0, 1);  // global_prefetch_b8
    } else {
        s = d = (int)(gw - E_);
    }
    float w = ex[gw] / den[d];
    const float* hs = h   + (long)s * HID;
    float*       od = out + (long)d * HID;
    atomicAdd(od + lane,      hs[lane]      * w);
    atomicAdd(od + lane + 32, hs[lane + 32] * w);
}

// out = elu(out + bias)
__global__ void bias_elu_kernel(float* __restrict__ out, const float* __restrict__ bias, long n64) {
    long t = (long)blockIdx.x * blockDim.x + threadIdx.x;
    if (t >= n64) return;
    int f = (int)(t & (HID - 1));
    float v = out[t] + bias[f];
    out[t] = v > 0.f ? v : expm1f(v);
}

// logits = h @ fc_w + fc_b
__global__ void fc_kernel(const float* __restrict__ h, const float* __restrict__ fw,
                          const float* __restrict__ fb, float* __restrict__ out, int n) {
    __shared__ float sw[HID];
    if (threadIdx.x < HID) sw[threadIdx.x] = fw[threadIdx.x];
    __syncthreads();
    int i = blockIdx.x * blockDim.x + threadIdx.x;
    if (i >= n) return;
    const float* hr = h + (long)i * HID;
    float s = 0.f;
#pragma unroll 8
    for (int f = 0; f < HID; ++f) s += hr[f] * sw[f];
    out[i] = s + fb[0];
}

// ---------------------------------------------------------------------------
extern "C" void kernel_launch(void* const* d_in, const int* in_sizes, int n_in,
                              void* d_out, int out_size, void* d_ws, size_t ws_size,
                              hipStream_t stream) {
    const float* x    = (const float*)d_in[0];
    const int*   ei   = (const int*)  d_in[1];
    const float* ea   = (const float*)d_in[2];
    const float* W1p  = (const float*)d_in[3];
    const float* as1  = (const float*)d_in[4];
    const float* ad1  = (const float*)d_in[5];
    const float* We1p = (const float*)d_in[6];
    const float* aev1 = (const float*)d_in[7];
    const float* b1   = (const float*)d_in[8];
    const float* W2p  = (const float*)d_in[9];
    const float* as2  = (const float*)d_in[10];
    const float* ad2  = (const float*)d_in[11];
    const float* We2p = (const float*)d_in[12];
    const float* aev2 = (const float*)d_in[13];
    const float* b2   = (const float*)d_in[14];
    const float* fcw  = (const float*)d_in[15];
    const float* fcb  = (const float*)d_in[16];
    (void)n_in; (void)out_size; (void)ws_size;

    const int n  = in_sizes[0] / HID;   // nodes
    const int E_ = in_sizes[1] / 2;     // edges
    const int* srcIdx = ei;
    const int* dstIdx = ei + E_;

    const int T = 256;
    auto blocks = [](long cnt, int t) { return (int)((cnt + t - 1) / t); };

    // -------- workspace carve-up (256B-aligned float regions) --------
    float* w = (float*)d_ws;
    size_t off = 0;
    auto carve = [&](size_t ne) { float* p = w + off; off += (ne + 63) & ~(size_t)63; return p; };
    float* h    = carve((size_t)n * HID);   // projected features (shared by both layers)
    float* o1   = carve((size_t)n * HID);   // layer-1 output
    float* o2   = carve((size_t)n * HID);   // layer-2 output
    float* aS   = carve((size_t)n);
    float* aD   = carve((size_t)n);
    float* ae1  = carve((size_t)E_);
    float* ae2  = carve((size_t)E_);
    float* al1  = carve((size_t)n);
    float* al2  = carve((size_t)n);
    float* lg   = carve((size_t)E_ + n);    // logits, then exp() in place
    float* smax = carve((size_t)n);
    float* den  = carve((size_t)n);
    float* deg  = carve((size_t)n);
    float* lsum = carve((size_t)n * FE);
    float* we1  = carve(64);
    float* we2  = carve(64);

    // -------- shared preprocessing (both layers) --------
    fill_kernel<<<blocks(n, T), T, 0, stream>>>(deg, n, 0.f);
    fill_kernel<<<blocks((long)n * FE, T), T, 0, stream>>>(lsum, (long)n * FE, 0.f);
    we_ae_kernel<<<1, 64, 0, stream>>>(We1p, aev1, We2p, aev2, we1, we2);
    deg_attr_kernel<<<blocks((long)E_ * FE, T), T, 0, stream>>>(dstIdx, ea, deg, lsum, E_);
    loop_alpha_kernel<<<blocks(n, T), T, 0, stream>>>(deg, lsum, we1, we2, al1, al2, n);
    alpha_e_edges_kernel<<<blocks(E_, T), T, 0, stream>>>(ea, we1, we2, ae1, ae2, E_);

    const int rowTiles = (n + 15) / 16;
    const int gemmBlocks = (rowTiles + 7) / 8;   // 8 row-stripes (waves) per block
    const long M = (long)E_ + n;

    auto run_layer = [&](const float* xin, const float* Wg,
                         const float* asv, const float* adv,
                         const float* aeE, const float* aeL,
                         const float* bias, float* outBuf) {
        fill_kernel<<<blocks((long)n * HID, T), T, 0, stream>>>(outBuf, (long)n * HID, 0.f);
        fill_kernel<<<blocks(n, T), T, 0, stream>>>(smax, n, -INFINITY);
        fill_kernel<<<blocks(n, T), T, 0, stream>>>(den, n, 0.f);
        gemm_bf16_wmma_kernel<<<gemmBlocks, T, 0, stream>>>(xin, Wg, h, n);
        alpha_nodes_kernel<<<blocks(n, T), T, 0, stream>>>(h, asv, adv, aS, aD, n);
        logits_kernel<<<blocks(M, T), T, 0, stream>>>(srcIdx, dstIdx, aS, aD, aeE, aeL, lg, smax, E_, n);
        exp_kernel<<<blocks(M, T), T, 0, stream>>>(dstIdx, lg, smax, den, E_, n);
        aggregate_kernel<<<blocks(M * 32, T), T, 0, stream>>>(srcIdx, dstIdx, h, lg, den, outBuf, E_, n);
        bias_elu_kernel<<<blocks((long)n * HID, T), T, 0, stream>>>(outBuf, bias, (long)n * HID);
    };

    run_layer(x,  W1p, as1, ad1, ae1, al1, b1, o1);
    run_layer(o1, W2p, as2, ad2, ae2, al2, b2, o2);
    fc_kernel<<<blocks(n, T), T, 0, stream>>>(o2, fcw, fcb, (float*)d_out, n);
}
